// GeATLayer_369367188029
// MI455X (gfx1250) — matrix operand order repaired and checked by reading, not compiled, not run on hardware
//
#include <hip/hip_runtime.h>

typedef __attribute__((ext_vector_type(16))) _Float16 v16h;
typedef __attribute__((ext_vector_type(8)))  _Float16 v8h;
typedef __attribute__((ext_vector_type(8)))  float    v8f;
typedef __attribute__((ext_vector_type(4)))  int      i32x4;
typedef __attribute__((ext_vector_type(2)))  int      i32x2;

#define N_ATOMS 2048
#define ATOM_D  64
#define NHEAD   8
#define DMODEL  512
#define NBOND   8
#define TI 16
#define TJ 32
#define NEG_SLOPE 0.2f

// ---------------- CDNA5 async global->LDS helpers --------------------
typedef __attribute__((address_space(1))) i32x4 g_i32x4;
typedef __attribute__((address_space(3))) i32x4 l_i32x4;
typedef __attribute__((address_space(1))) i32x2 g_i32x2;
typedef __attribute__((address_space(3))) i32x2 l_i32x2;

template <int OFF>
__device__ __forceinline__ void async_b128(const void* g, void* l) {
#if __has_builtin(__builtin_amdgcn_global_load_async_to_lds_b128)
  __builtin_amdgcn_global_load_async_to_lds_b128(
      (g_i32x4*)(unsigned long long)g,
      (l_i32x4*)(unsigned int)(unsigned long long)l,   // low 32 bits = LDS offset
      OFF, 0);
#else
  *(uint4*)((char*)l + OFF) = *(const uint4*)((const char*)g + OFF);
#endif
}
template <int OFF>
__device__ __forceinline__ void async_b64(const void* g, void* l) {
#if __has_builtin(__builtin_amdgcn_global_load_async_to_lds_b64)
  __builtin_amdgcn_global_load_async_to_lds_b64(
      (g_i32x2*)(unsigned long long)g,
      (l_i32x2*)(unsigned int)(unsigned long long)l,
      OFF, 0);
#else
  *(uint2*)((char*)l + OFF) = *(const uint2*)((const char*)g + OFF);
#endif
}
__device__ __forceinline__ void wait_async() {
#if __has_builtin(__builtin_amdgcn_global_load_async_to_lds_b128)
#if __has_builtin(__builtin_amdgcn_s_wait_asynccnt)
  __builtin_amdgcn_s_wait_asynccnt(0);
#else
  asm volatile("s_wait_asynccnt 0x0" ::: "memory");
#endif
#endif
}

__device__ __forceinline__ float lane_bcast(float v, int srcLane) {
  return __builtin_bit_cast(float,
      __builtin_amdgcn_ds_bpermute(srcLane << 2, __builtin_bit_cast(int, v)));
}

// ------------------------------------------------------------------
// Kernel 1: fold Qw/Kw into per-(bond,head) 64x64 matrices
//   W1[d,c] = sum_dd Qw[d,dd]*a[k,dd,h], c=k*8+h  (b1 from Qb; W2/b2 from Kw/Kb)
// Also pre-transpose Pw to f16: Pwt[n,k] = Pw[k,n]  (64 x 512)
// ------------------------------------------------------------------
__global__ __launch_bounds__(256) void prep_kernel(
    const float* __restrict__ Qw, const float* __restrict__ Qb,
    const float* __restrict__ Kw, const float* __restrict__ Kb,
    const float* __restrict__ a,  const float* __restrict__ Pw,
    float* __restrict__ W1, float* __restrict__ b1,
    float* __restrict__ W2, float* __restrict__ b2,
    _Float16* __restrict__ Pwt)
{
  for (int e = threadIdx.x; e < ATOM_D * NBOND * NHEAD; e += 256) {
    int d = e >> 6, c = e & 63;
    int k = c >> 3, h = c & 7;
    const float* qrow = Qw + d * DMODEL;
    const float* krow = Kw + d * DMODEL;
    const float* ak = a + k * (2 * DMODEL * NHEAD) + h;
    float s1 = 0.f, s2 = 0.f;
    for (int dd = 0; dd < DMODEL; ++dd) {
      s1 += qrow[dd] * ak[dd * NHEAD];
      s2 += krow[dd] * ak[(DMODEL + dd) * NHEAD];
    }
    W1[d * 64 + c] = s1;
    W2[d * 64 + c] = s2;
  }
  for (int c = threadIdx.x; c < 64; c += 256) {
    int k = c >> 3, h = c & 7;
    const float* ak = a + k * (2 * DMODEL * NHEAD) + h;
    float s1 = 0.f, s2 = 0.f;
    for (int dd = 0; dd < DMODEL; ++dd) {
      s1 += Qb[dd] * ak[dd * NHEAD];
      s2 += Kb[dd] * ak[(DMODEL + dd) * NHEAD];
    }
    b1[c] = s1;
    b2[c] = s2;
  }
  for (int u = threadIdx.x; u < 64 * DMODEL; u += 256) {
    int n = u >> 9, k = u & (DMODEL - 1);
    Pwt[n * DMODEL + k] = (_Float16)Pw[k * 64 + n];
  }
}

// ------------------------------------------------------------------
// Kernel 2: S1 (2048x64 f32), S2 (2048x64 f32), and V converted to f16
// directly in the transposed per-tile layout Vt[tile][col][j&31].
// ------------------------------------------------------------------
__global__ __launch_bounds__(256) void proj_kernel(
    const float* __restrict__ X,
    const float* __restrict__ W1, const float* __restrict__ b1,
    const float* __restrict__ W2, const float* __restrict__ b2,
    const float* __restrict__ Vw, const float* __restrict__ Vb,
    float* __restrict__ S1, float* __restrict__ S2, _Float16* __restrict__ Vt)
{
  __shared__ float x[ATOM_D];
  int i = blockIdx.x;
  if (threadIdx.x < ATOM_D) x[threadIdx.x] = X[i * ATOM_D + threadIdx.x];
  __syncthreads();
  for (int o = threadIdx.x; o < 128 + DMODEL; o += 256) {
    if (o < 64) {
      float s = b1[o];
      #pragma unroll 8
      for (int d = 0; d < ATOM_D; ++d) s += x[d] * W1[d * 64 + o];
      S1[i * 64 + o] = s;
    } else if (o < 128) {
      int c = o - 64;
      float s = b2[c];
      #pragma unroll 8
      for (int d = 0; d < ATOM_D; ++d) s += x[d] * W2[d * 64 + c];
      S2[i * 64 + c] = s;
    } else {
      int c = o - 128;
      float s = Vb[c];
      #pragma unroll 8
      for (int d = 0; d < ATOM_D; ++d) s += x[d] * Vw[d * DMODEL + c];
      Vt[((i >> 5) * DMODEL + c) * TJ + (i & 31)] = (_Float16)s;
    }
  }
}

// ------------------------------------------------------------------
// Kernel 3: streaming (flash-style) edge-typed attention + output proj.
// Block = 16 rows x all heads; 8 wave32s, wave w == head w.
// All tile staging is async global->LDS (ASYNCcnt) block copies.
// ------------------------------------------------------------------
struct Phase1 {
  float    s1[TI * ATOM_D];   // 4 KB  : S1 rows for i-block (contiguous in global)
  float    s2[TJ * ATOM_D];   // 8 KB  : S2 rows for j-tile  (contiguous in global)
  int      ed[TI * TJ];       // 2 KB  : edges tile (row-wise b64 copies)
  _Float16 vt[DMODEL * TJ];   // 32 KB : V tile, pre-transposed (contiguous in global)
};
struct Phase2 {
  _Float16 outh[TI * DMODEL];        // 16 KB : normalized attention output (f16)
  float    partial[NHEAD * TI * 64]; // 32 KB : per-head projection partials
};
union Smem { Phase1 p1; Phase2 p2; };

__global__ __launch_bounds__(256) void attn_kernel(
    const int*      __restrict__ edges,
    const float*    __restrict__ S1,
    const float*    __restrict__ S2,
    const _Float16* __restrict__ Vt,
    const _Float16* __restrict__ Pwt,
    const float*    __restrict__ Pb,
    float*          __restrict__ out)
{
  __shared__ __attribute__((aligned(16))) Smem sm;
  const int t    = threadIdx.x;
  const int lane = t & 31;
  const int h    = t >> 5;          // head == wave id
  const int half = lane >> 4;       // 0: K-low half, 1: K-high half
  const int mrow = lane & 15;       // A-fragment row / B,C column
  const int i0   = blockIdx.x * TI;

  // S1 i-block: one contiguous 4KB async copy (16B per thread)
  async_b128<0>((const char*)(S1 + i0 * ATOM_D) + t * 16, (char*)sm.p1.s1 + t * 16);

  v8f acc[4] = {};                  // 16x64 f32 accumulator (head slice)
  float m = -__builtin_inff();
  float l = 0.f;

  for (int j0 = 0; j0 < N_ATOMS; j0 += TJ) {
    __syncthreads();   // previous tile fully consumed -> safe to overwrite LDS

    // edges tile: 16 rows x 128B; one b64 per thread
    {
      const int* gsrc = edges + (i0 + (t >> 4)) * N_ATOMS + j0 + ((t & 15) * 2);
      async_b64<0>(gsrc, sm.p1.ed + t * 2);
    }
    // S2 tile: contiguous 8KB; two b128 per thread
    {
      const char* gs = (const char*)(S2 + (size_t)j0 * ATOM_D) + t * 16;
      char* ls = (char*)sm.p1.s2 + t * 16;
      async_b128<0>(gs, ls);
      async_b128<4096>(gs, ls);
    }
    // V tile (pre-transposed f16): contiguous 32KB; eight b128 per thread
    {
      const char* gv = (const char*)(Vt + (size_t)(j0 >> 5) * (DMODEL * TJ)) + t * 16;
      char* lv = (char*)sm.p1.vt + t * 16;
      async_b128<0>(gv, lv);      async_b128<4096>(gv, lv);
      async_b128<8192>(gv, lv);   async_b128<12288>(gv, lv);
      async_b128<16384>(gv, lv);  async_b128<20480>(gv, lv);
      async_b128<24576>(gv, lv);  async_b128<28672>(gv, lv);
    }
    wait_async();
    __syncthreads();

    // 16 logits per lane, laid out exactly as the f16 A-fragment demands:
    // idx<8 : K = half*8+idx ; idx>=8 : K = 16 + half*8 + (idx-8)
    float ev[16];
    #pragma unroll
    for (int idx = 0; idx < 16; ++idx) {
      int jl = ((idx & 8) ? 16 : 0) + half * 8 + (idx & 7);
      int ec = sm.p1.ed[mrow * TJ + jl];
      int k  = ec < 0 ? 0 : (ec > NBOND - 1 ? NBOND - 1 : ec);
      int c  = k * NHEAD + h;
      float val = sm.p1.s1[mrow * ATOM_D + c] + sm.p1.s2[jl * ATOM_D + c];
      val = val >= 0.f ? val : NEG_SLOPE * val;
      ev[idx] = (ec >= 0) ? val : 0.f;   // masked entries contribute e=0
    }

    // online softmax: row stats live symmetrically in lanes L and L^16
    float mt = ev[0];
    #pragma unroll
    for (int idx = 1; idx < 16; ++idx) mt = fmaxf(mt, ev[idx]);
    mt = fmaxf(mt, lane_bcast(mt, lane ^ 16));
    float mnew  = fmaxf(m, mt);
    float scale = __expf(m - mnew);
    m = mnew;

    float ps = 0.f;
    v16h af;
    #pragma unroll
    for (int idx = 0; idx < 16; ++idx) {
      float p = __expf(ev[idx] - mnew);
      ps += p;
      af[idx] = (_Float16)p;
    }
    ps += lane_bcast(ps, lane ^ 16);
    l = l * scale + ps;

    // rescale accumulators: C-layout row = half*8 + vgpr-index
    #pragma unroll
    for (int vi = 0; vi < 8; ++vi) {
      float rs = lane_bcast(scale, half * 8 + vi);
      #pragma unroll
      for (int n = 0; n < 4; ++n) acc[n][vi] *= rs;
    }

    // B fragments from transposed V tile (two contiguous ds_load_b128 each)
    #pragma unroll
    for (int n = 0; n < 4; ++n) {
      int col = h * 64 + n * 16 + mrow;
      const v8h* bp = (const v8h*)&sm.p1.vt[col * TJ + half * 8];
      v8h blo = bp[0], bhi = bp[2];
      v16h bf;
      #pragma unroll
      for (int q = 0; q < 8; ++q) { bf[q] = blo[q]; bf[8 + q] = bhi[q]; }
      acc[n] = __builtin_amdgcn_wmma_f32_16x16x32_f16(
          false, af, false, bf, (short)0, acc[n], false, false);
    }
  }

  __syncthreads();   // done reading phase-1 LDS

  // normalize by l and stash head slice as f16 (A operand of final GEMM)
  #pragma unroll
  for (int vi = 0; vi < 8; ++vi) {
    float rl  = lane_bcast(l, half * 8 + vi);
    float inv = 1.f / rl;
    int row = half * 8 + vi;
    #pragma unroll
    for (int n = 0; n < 4; ++n) {
      int col = h * 64 + n * 16 + mrow;
      sm.p2.outh[row * DMODEL + col] = (_Float16)(acc[n][vi] * inv);
    }
  }
  __syncthreads();

  // final projection: res = outh @ Pw. Each wave does its 64-wide K slice.
  // B fragments come from pre-transposed f16 Pwt[n, k] (contiguous in K).
  v8f c2[4] = {};
  #pragma unroll
  for (int kb = 0; kb < 64; kb += 32) {
    const _Float16* arow = &sm.p2.outh[mrow * DMODEL + h * 64 + kb];
    v8h alo = *(const v8h*)(arow + half * 8);
    v8h ahi = *(const v8h*)(arow + 16 + half * 8);
    v16h af;
    #pragma unroll
    for (int q = 0; q < 8; ++q) { af[q] = alo[q]; af[8 + q] = ahi[q]; }
    #pragma unroll
    for (int n = 0; n < 4; ++n) {
      const _Float16* brow = Pwt + (size_t)(n * 16 + mrow) * DMODEL + h * 64 + kb + half * 8;
      v8h blo = *(const v8h*)brow;
      v8h bhi = *(const v8h*)(brow + 16);
      v16h bf;
      #pragma unroll
      for (int q = 0; q < 8; ++q) { bf[q] = blo[q]; bf[8 + q] = bhi[q]; }
      c2[n] = __builtin_amdgcn_wmma_f32_16x16x32_f16(
          false, af, false, bf, (short)0, c2[n], false, false);
    }
  }

  // per-head partials -> LDS, cross-head reduction, +Pb, store
  #pragma unroll
  for (int vi = 0; vi < 8; ++vi) {
    int row = half * 8 + vi;
    #pragma unroll
    for (int n = 0; n < 4; ++n)
      sm.p2.partial[(h * TI + row) * ATOM_D + n * 16 + mrow] = c2[n][vi];
  }
  __syncthreads();
  for (int u = t; u < TI * ATOM_D; u += 256) {
    int row = u >> 6, c = u & 63;
    float s = Pb[c];
    #pragma unroll
    for (int hh = 0; hh < NHEAD; ++hh)
      s += sm.p2.partial[(hh * TI + row) * ATOM_D + c];
    out[(i0 + row) * ATOM_D + c] = s;
  }
}

extern "C" void kernel_launch(void* const* d_in, const int* in_sizes, int n_in,
                              void* d_out, int out_size, void* d_ws, size_t ws_size,
                              hipStream_t stream) {
  const float* X     = (const float*)d_in[0];
  const int*   edges = (const int*)d_in[1];
  const float* Qw    = (const float*)d_in[2];
  const float* Qb    = (const float*)d_in[3];
  const float* Kw    = (const float*)d_in[4];
  const float* Kb    = (const float*)d_in[5];
  const float* Vw    = (const float*)d_in[6];
  const float* Vb    = (const float*)d_in[7];
  const float* a     = (const float*)d_in[8];
  const float* Pw    = (const float*)d_in[9];
  const float* Pb    = (const float*)d_in[10];
  (void)in_sizes; (void)n_in; (void)out_size; (void)ws_size;

  float* ws = (float*)d_ws;
  float* W1 = ws;                         // 64*64
  float* b1 = W1 + 64 * 64;               // 64
  float* W2 = b1 + 64;                    // 64*64
  float* b2 = W2 + 64 * 64;               // 64
  float* S1 = b2 + 64;                    // 2048*64
  float* S2 = S1 + N_ATOMS * 64;          // 2048*64
  _Float16* Vt  = (_Float16*)(S2 + N_ATOMS * 64);      // 2048*512 f16 (2 MB)
  _Float16* Pwt = Vt + (size_t)N_ATOMS * DMODEL;       // 64*512 f16
  float* out = (float*)d_out;

  prep_kernel<<<1, 256, 0, stream>>>(Qw, Qb, Kw, Kb, a, Pw, W1, b1, W2, b2, Pwt);
  proj_kernel<<<N_ATOMS, 256, 0, stream>>>(X, W1, b1, W2, b2, Vw, Vb, S1, S2, Vt);
  attn_kernel<<<N_ATOMS / TI, 256, 0, stream>>>(edges, S1, S2, Vt, Pwt, Pb, out);
}
